// MoleculeMPNN_69904887710203
// MI455X (gfx1250) — matrix operand order repaired and checked by reading, not compiled, not run on hardware
//
#include <hip/hip_runtime.h>
#include <hip/hip_bf16.h>

// ---------------------------------------------------------------------------
// MoleculeMPNN forward for MI455X (gfx1250, wave32, WMMA).
// bf16 storage for the two big tensors (w = E x 1024, feats = C x 160) and
// bf16 WMMA (v_wmma_f32_16x16x32_bf16) for the en3 and fc0 GEMMs; fp32
// accumulation everywhere else. fc1 + its BatchNorm are folded into
// per-column scales so fc0->relu->bn->fc1 is fused in registers.
// ---------------------------------------------------------------------------

typedef __bf16 bf16;
typedef __attribute__((ext_vector_type(16))) __bf16 v16bf;
typedef __attribute__((ext_vector_type(8)))  float  v8f;

#define DEV static __device__ __forceinline__

constexpr int N_ = 30000, E_ = 120000, H_ = 32, NF_ = 32, EF_ = 8;
constexpr int B_ = 512, C_ = 100000, STEPS_ = 6;
constexpr float EPS_ = 1e-5f;

// ------------------------- WMMA tile loaders -------------------------------
// A (16x32 bf16, M x K): lanes 0-15 row M=lane, K = k0+{0..7,16..23};
// lanes 16-31 row M=lane-16, K = k0+{8..15,24..31}.
DEV v16bf load_A(const bf16* A, int lda, int m0, int k0, int lane) {
  int r  = lane & 15;
  int kb = k0 + ((lane & 16) ? 8 : 0);
  const bf16* p = A + (size_t)(m0 + r) * lda + kb;
  v16bf v;
#pragma unroll
  for (int e = 0; e < 8; ++e) { v[e] = p[e]; v[8 + e] = p[16 + e]; }
  return v;
}

// B (32x16 bf16, K x N) sourced from row-major weight W[n][k]:
// lane n=lane&15 holds K = k0+{0..7,16..23} (lanes 0-15) or k0+{8..15,24..31} (16-31)
// of row (n0+n) of W.
DEV v16bf load_Bt(const bf16* W, int ldw, int n0, int k0, int lane) {
  int n  = lane & 15;
  int kb = k0 + ((lane & 16) ? 8 : 0);
  const bf16* p = W + (size_t)(n0 + n) * ldw + kb;
  v16bf v;
#pragma unroll
  for (int e = 0; e < 8; ++e) { v[e] = p[e]; v[8 + e] = p[16 + e]; }
  return v;
}

DEV float sigmoidf(float x) { return 1.f / (1.f + expf(-x)); }

// monotonic float<->uint encoding for atomicMax on floats
DEV unsigned fenc(float f) {
  int i = __float_as_int(f);
  return (i < 0) ? ~(unsigned)i : ((unsigned)i | 0x80000000u);
}
DEV float fdec(unsigned u) {
  return (u & 0x80000000u) ? __int_as_float((int)(u & 0x7fffffffu))
                           : __int_as_float((int)~u);
}

// ------------------------------- kernels -----------------------------------

__global__ void k_zero(float* p, long n) {
  long i = (long)blockIdx.x * blockDim.x + threadIdx.x;
  if (i < n) p[i] = 0.f;
}

__global__ void k_prep_w3(const float* W, bf16* out) {          // 1024x64
  int i = blockIdx.x * 256 + threadIdx.x;
  if (i < 1024 * 64) out[i] = (bf16)W[i];
}

__global__ void k_prep_fc0(const float* W, bf16* out) {         // (1024,136)->(1024,160)
  int i = blockIdx.x * 256 + threadIdx.x;
  if (i >= 1024 * 160) return;
  int o = i / 160, k = i % 160;
  out[i] = (k < 136) ? (bf16)W[o * 136 + k] : (bf16)0.0f;
}

// Fold fc1's input BatchNorm into fc1: W1p[j][o] = W1[j][o]*s[o],
// const1[j] = sum_o (b[o]-rm[o]*s[o])*W1[j][o] + lb[j].
__global__ void k_prep_fc1(const float* W, const float* rm, const float* rv,
                           const float* g, const float* b, const float* lb,
                           float* W1p, float* const1) {
  int j = blockIdx.x, t = threadIdx.x;
  __shared__ float red[256];
  float part = 0.f;
  for (int o = t; o < 1024; o += 256) {
    float s  = g[o] * rsqrtf(rv[o] + EPS_);
    float tt = b[o] - rm[o] * s;
    float wv = W[j * 1024 + o];
    W1p[j * 1024 + o] = wv * s;
    part += tt * wv;
  }
  red[t] = part; __syncthreads();
  for (int s = 128; s > 0; s >>= 1) { if (t < s) red[t] += red[t + s]; __syncthreads(); }
  if (t == 0) const1[j] = red[0] + lb[j];
}

// x = relu(bn_lin(bn_lin(node, emb0, relu), emb1, none)); one wave per node
__global__ void k_node_embed(const float* __restrict__ node,
    const float* rm0, const float* rv0, const float* g0, const float* b0,
    const float* W0, const float* lb0,
    const float* rm1, const float* rv1, const float* g1, const float* b1,
    const float* W1, const float* lb1, float* __restrict__ state) {
  int n = (blockIdx.x * blockDim.x + threadIdx.x) >> 5;
  int l = threadIdx.x & 31;
  if (n >= N_) return;
  float x = node[n * NF_ + l];
  float v = (x - rm0[l]) * rsqrtf(rv0[l] + EPS_) * g0[l] + b0[l];
  int j0 = l, j1 = l + 32;
  float a0 = lb0[j0], a1 = lb0[j1];
#pragma unroll
  for (int i = 0; i < 32; ++i) {
    float vi = __shfl(v, i, 32);
    a0 += vi * W0[j0 * 32 + i];
    a1 += vi * W0[j1 * 32 + i];
  }
  a0 = fmaxf(a0, 0.f); a1 = fmaxf(a1, 0.f);
  a0 = (a0 - rm1[j0]) * rsqrtf(rv1[j0] + EPS_) * g1[j0] + b1[j0];
  a1 = (a1 - rm1[j1]) * rsqrtf(rv1[j1] + EPS_) * g1[j1] + b1[j1];
  float acc = lb1[l];
#pragma unroll
  for (int j = 0; j < 32; ++j)
    acc += __shfl(a0, j, 32) * W1[l * 64 + j] + __shfl(a1, j, 32) * W1[l * 64 + 32 + j];
  state[n * H_ + l] = fmaxf(acc, 0.f);
}

struct EdgeNetParams {
  const float *rm0,*rv0,*g0,*b0,*W0,*lb0;   // en0: 8->32
  const float *rm1,*rv1,*g1,*b1,*W1,*lb1;   // en1: 32->64
  const float *rm2,*rv2,*g2,*b2,*W2,*lb2;   // en2: 64->64
  const float *rm3,*rv3,*g3,*b3;            // en3 BN (folded into e3n)
};

// edge -> en0 -> en1 -> en2 -> BN(en3) -> e3n (bf16, E x 64); one wave per edge
__global__ void k_edge_net(const float* __restrict__ edge, EdgeNetParams P,
                           bf16* __restrict__ e3n) {
  int e = (blockIdx.x * blockDim.x + threadIdx.x) >> 5;
  int l = threadIdx.x & 31;
  if (e >= E_) return;
  float v = 0.f;
  if (l < EF_) {
    float x = edge[e * EF_ + l];
    v = (x - P.rm0[l]) * rsqrtf(P.rv0[l] + EPS_) * P.g0[l] + P.b0[l];
  }
  float t = P.lb0[l];
#pragma unroll
  for (int i = 0; i < 8; ++i) t += __shfl(v, i, 32) * P.W0[l * 8 + i];
  t = fmaxf(t, 0.f);
  t = (t - P.rm1[l]) * rsqrtf(P.rv1[l] + EPS_) * P.g1[l] + P.b1[l];
  int j0 = l, j1 = l + 32;
  float u0 = P.lb1[j0], u1 = P.lb1[j1];
#pragma unroll
  for (int i = 0; i < 32; ++i) {
    float ti = __shfl(t, i, 32);
    u0 += ti * P.W1[j0 * 32 + i];
    u1 += ti * P.W1[j1 * 32 + i];
  }
  u0 = fmaxf(u0, 0.f); u1 = fmaxf(u1, 0.f);
  u0 = (u0 - P.rm2[j0]) * rsqrtf(P.rv2[j0] + EPS_) * P.g2[j0] + P.b2[j0];
  u1 = (u1 - P.rm2[j1]) * rsqrtf(P.rv2[j1] + EPS_) * P.g2[j1] + P.b2[j1];
  float s0 = P.lb2[j0], s1 = P.lb2[j1];
#pragma unroll
  for (int i = 0; i < 32; ++i) {
    float a = __shfl(u0, i, 32), bb = __shfl(u1, i, 32);
    s0 += a * P.W2[j0 * 64 + i] + bb * P.W2[j0 * 64 + 32 + i];
    s1 += a * P.W2[j1 * 64 + i] + bb * P.W2[j1 * 64 + 32 + i];
  }
  s0 = fmaxf(s0, 0.f); s1 = fmaxf(s1, 0.f);
  s0 = (s0 - P.rm3[j0]) * rsqrtf(P.rv3[j0] + EPS_) * P.g3[j0] + P.b3[j0];
  s1 = (s1 - P.rm3[j1]) * rsqrtf(P.rv3[j1] + EPS_) * P.g3[j1] + P.b3[j1];
  e3n[(size_t)e * 64 + j0] = (bf16)s0;
  e3n[(size_t)e * 64 + j1] = (bf16)s1;
}

// w = e3n @ en3.W^T + lb  (M=E, K=64, N=1024) -> bf16. 8 waves/block, each
// wave 16 rows x 128 cols via v_wmma_f32_16x16x32_bf16.
__global__ void k_gemm_en3(const bf16* __restrict__ A, const bf16* __restrict__ Wb,
                           const float* __restrict__ lb, bf16* __restrict__ wout) {
  int wv = threadIdx.x >> 5, lane = threadIdx.x & 31;
  int m0 = blockIdx.x * 16;
  v16bf a0 = load_A(A, 64, m0, 0, lane);
  v16bf a1 = load_A(A, 64, m0, 32, lane);
  int n = lane & 15, mr = (lane & 16) ? 8 : 0;
#pragma unroll
  for (int nt = 0; nt < 8; ++nt) {
    int n0 = wv * 128 + nt * 16;
    v16bf b0 = load_Bt(Wb, 64, n0, 0, lane);
    v16bf b1 = load_Bt(Wb, 64, n0, 32, lane);
    v8f acc = {};
    acc = __builtin_amdgcn_wmma_f32_16x16x32_bf16(false, a0, false, b0, (short)0, acc, false, false);
    acc = __builtin_amdgcn_wmma_f32_16x16x32_bf16(false, a1, false, b1, (short)0, acc, false, false);
    float bias = lb[n0 + n];
#pragma unroll
    for (int r = 0; r < 8; ++r)
      wout[(size_t)(m0 + mr + r) * 1024 + n0 + n] = (bf16)(acc[r] + bias);
  }
}

__global__ void k_deg(const int* __restrict__ eidx, float* __restrict__ deg) {
  int e = blockIdx.x * blockDim.x + threadIdx.x;
  if (e >= E_) return;
  atomicAdd(&deg[eidx[e * 2 + 1]], 1.f);
}

// msg[e][o] = sum_i state[src][i]*w[e][i*32+o]; scatter-add into agg[dst]
__global__ void k_msg(const float* __restrict__ state, const bf16* __restrict__ wbuf,
                      const int* __restrict__ eidx, float* __restrict__ agg) {
  int e = (blockIdx.x * blockDim.x + threadIdx.x) >> 5;
  int o = threadIdx.x & 31;
  if (e >= E_) return;
  int src = eidx[e * 2 + 0], dst = eidx[e * 2 + 1];
  float a = state[src * 32 + o];
  const bf16* wp = wbuf + (size_t)e * 1024 + o;
  float acc = 0.f;
#pragma unroll
  for (int i = 0; i < 32; ++i)
    acc += __shfl(a, i, 32) * (float)wp[i * 32];
  atomicAdd(&agg[dst * 32 + o], acc);
}

// m = relu(state@root + agg/deg + nn_bias); GRU(state, m) -> state
__global__ void k_gru(float* __restrict__ state, const float* __restrict__ agg,
                      const float* __restrict__ deg, const float* __restrict__ root,
                      const float* __restrict__ nbias,
                      const float* wih, const float* whh,
                      const float* bih, const float* bhh) {
  int n = (blockIdx.x * blockDim.x + threadIdx.x) >> 5;
  int o = threadIdx.x & 31;
  if (n >= N_) return;
  float h = state[n * 32 + o];
  float m = nbias[o] + agg[n * 32 + o] / fmaxf(deg[n], 1.f);
#pragma unroll
  for (int j = 0; j < 32; ++j) m += __shfl(h, j, 32) * root[j * 32 + o];
  m = fmaxf(m, 0.f);
  float gr = bih[o], gz = bih[o + 32], gn = bih[o + 64];
  float hr = bhh[o], hz = bhh[o + 32], hn = bhh[o + 64];
#pragma unroll
  for (int j = 0; j < 32; ++j) {
    float mj = __shfl(m, j, 32), hj = __shfl(h, j, 32);
    gr += mj * wih[o * 32 + j];          hr += hj * whh[o * 32 + j];
    gz += mj * wih[(o + 32) * 32 + j];   hz += hj * whh[(o + 32) * 32 + j];
    gn += mj * wih[(o + 64) * 32 + j];   hn += hj * whh[(o + 64) * 32 + j];
  }
  float r  = sigmoidf(gr + hr);
  float z  = sigmoidf(gz + hz);
  float nn = tanhf(gn + r * hn);
  state[n * 32 + o] = (1.f - z) * nn + z * h;
}

// Set2Set LSTM cell; one wave per batch element. Writes q (=hs) into q_star[:,:32].
__global__ void k_lstm(float* __restrict__ q_star, float* __restrict__ hs,
                       float* __restrict__ cs, const float* wih, const float* whh,
                       const float* bih, const float* bhh) {
  int b = (blockIdx.x * blockDim.x + threadIdx.x) >> 5;
  int l = threadIdx.x & 31;
  if (b >= B_) return;
  float q0 = q_star[b * 64 + l], q1 = q_star[b * 64 + 32 + l];
  float h  = hs[b * 32 + l];
  float gi = bih[l] + bhh[l],           gf = bih[l + 32] + bhh[l + 32];
  float gg = bih[l + 64] + bhh[l + 64], go = bih[l + 96] + bhh[l + 96];
#pragma unroll
  for (int j = 0; j < 32; ++j) {
    float a0 = __shfl(q0, j, 32), a1 = __shfl(q1, j, 32), hh = __shfl(h, j, 32);
    gi += a0 * wih[l * 64 + j]          + a1 * wih[l * 64 + 32 + j]          + hh * whh[l * 32 + j];
    gf += a0 * wih[(l + 32) * 64 + j]   + a1 * wih[(l + 32) * 64 + 32 + j]   + hh * whh[(l + 32) * 32 + j];
    gg += a0 * wih[(l + 64) * 64 + j]   + a1 * wih[(l + 64) * 64 + 32 + j]   + hh * whh[(l + 64) * 32 + j];
    go += a0 * wih[(l + 96) * 64 + j]   + a1 * wih[(l + 96) * 64 + 32 + j]   + hh * whh[(l + 96) * 32 + j];
  }
  float c    = sigmoidf(gf) * cs[b * 32 + l] + sigmoidf(gi) * tanhf(gg);
  float hnew = sigmoidf(go) * tanhf(c);
  cs[b * 32 + l] = c;
  hs[b * 32 + l] = hnew;
  q_star[b * 64 + l] = hnew;
}

__global__ void k_s2s_reset(float* __restrict__ q_star, unsigned* __restrict__ emax,
                            float* __restrict__ denom) {
  int i = blockIdx.x * blockDim.x + threadIdx.x;
  if (i >= B_ * 32) return;
  int b = i >> 5, o = i & 31;
  q_star[b * 64 + 32 + o] = 0.f;
  if (o == 0) { emax[b] = 0u; denom[b] = 0.f; }
}

__global__ void k_attn_e(const float* __restrict__ state, const float* __restrict__ hs,
                         const int* __restrict__ bidx, float* __restrict__ ebuf,
                         unsigned* __restrict__ emax) {
  int n = (blockIdx.x * blockDim.x + threadIdx.x) >> 5;
  int l = threadIdx.x & 31;
  if (n >= N_) return;
  int b = bidx[n];
  float p = state[n * 32 + l] * hs[b * 32 + l];
#pragma unroll
  for (int m = 16; m > 0; m >>= 1) p += __shfl_xor(p, m, 32);
  if (l == 0) { ebuf[n] = p; atomicMax(&emax[b], fenc(p)); }
}

__global__ void k_attn_a(const float* __restrict__ ebuf, const unsigned* __restrict__ emax,
                         const int* __restrict__ bidx, float* __restrict__ anum,
                         float* __restrict__ denom) {
  int n = blockIdx.x * blockDim.x + threadIdx.x;
  if (n >= N_) return;
  int b = bidx[n];
  float a = expf(ebuf[n] - fdec(emax[b]));
  anum[n] = a;
  atomicAdd(&denom[b], a);
}

__global__ void k_attn_r(const float* __restrict__ anum, const float* __restrict__ denom,
                         const float* __restrict__ state, const int* __restrict__ bidx,
                         float* __restrict__ q_star) {
  int n = (blockIdx.x * blockDim.x + threadIdx.x) >> 5;
  int o = threadIdx.x & 31;
  if (n >= N_) return;
  int b = bidx[n];
  float coef = anum[n] / denom[b];
  atomicAdd(&q_star[b * 64 + 32 + o], coef * state[n * 32 + o]);
}

// Gather + BN-normalize feats (C x 160 bf16, K-padded with zeros)
__global__ void k_feats(const float* __restrict__ q_star, const float* __restrict__ state,
                        const int* __restrict__ cbidx, const int* __restrict__ cidx,
                        const int* __restrict__ ctype,
                        const float* rm, const float* rv, const float* g, const float* bb,
                        bf16* __restrict__ featsA) {
  long i = (long)blockIdx.x * blockDim.x + threadIdx.x;
  if (i >= (long)C_ * 160) return;
  int c = (int)(i / 160), k = (int)(i % 160);
  float raw;
  if (k < 64)       raw = q_star[cbidx[c] * 64 + k];
  else if (k < 96)  raw = state[cidx[c * 2 + 0] * 32 + (k - 64)];
  else if (k < 128) raw = state[cidx[c * 2 + 1] * 32 + (k - 96)];
  else if (k < 136) raw = (float)ctype[c * 8 + (k - 128)];
  else { featsA[i] = (bf16)0.f; return; }
  float s = g[k] * rsqrtf(rv[k] + EPS_);
  featsA[i] = (bf16)((raw - rm[k]) * s + bb[k]);
}

// fc0 (WMMA, M=C K=160 N=1024) fused with relu + folded fc1 -> out8 (C x 8)
__global__ void k_fc_wmma(const bf16* __restrict__ A, const bf16* __restrict__ Wb,
                          const float* __restrict__ lb0, const float* __restrict__ W1p,
                          float* __restrict__ out8) {
  int wv = threadIdx.x >> 5, lane = threadIdx.x & 31;
  int c0 = blockIdx.x * 16;
  v16bf a[5];
#pragma unroll
  for (int ck = 0; ck < 5; ++ck) a[ck] = load_A(A, 160, c0, ck * 32, lane);
  int nn = lane & 15, mr = (lane & 16) ? 8 : 0;
  float outp[64];
#pragma unroll
  for (int i = 0; i < 64; ++i) outp[i] = 0.f;
#pragma unroll
  for (int nt = 0; nt < 8; ++nt) {
    int n0 = wv * 128 + nt * 16;
    v8f acc = {};
#pragma unroll
    for (int ck = 0; ck < 5; ++ck) {
      v16bf b = load_Bt(Wb, 160, n0, ck * 32, lane);
      acc = __builtin_amdgcn_wmma_f32_16x16x32_bf16(false, a[ck], false, b, (short)0, acc, false, false);
    }
    int col = n0 + nn;
    float bias = lb0[col];
    float wj[8];
#pragma unroll
    for (int j = 0; j < 8; ++j) wj[j] = W1p[j * 1024 + col];
#pragma unroll
    for (int r = 0; r < 8; ++r) {
      float h = fmaxf(acc[r] + bias, 0.f);
#pragma unroll
      for (int j = 0; j < 8; ++j) outp[r * 8 + j] += h * wj[j];
    }
  }
#pragma unroll
  for (int m = 1; m < 16; m <<= 1)
#pragma unroll
    for (int i = 0; i < 64; ++i) outp[i] += __shfl_xor(outp[i], m, 32);
  if ((lane & 15) < 8) {
    int j = lane & 7;
#pragma unroll
    for (int r = 0; r < 8; ++r)
      atomicAdd(&out8[(size_t)(c0 + mr + r) * 8 + j], outp[r * 8 + j]);
  }
}

__global__ void k_final(const float* __restrict__ out8, const int* __restrict__ ctb,
                        const float* __restrict__ const1, float* __restrict__ out) {
  int c = blockIdx.x * blockDim.x + threadIdx.x;
  if (c >= C_) return;
  int j = ctb[c];
  out[c] = out8[(size_t)c * 8 + j] + const1[j];
}

// ------------------------------ host side ----------------------------------

// d_in flat order: setup_inputs() dict insertion order, params depth-first.
enum {
  IN_NODE = 0, IN_EDGE, IN_EIDX, IN_NBIDX, IN_CIDX, IN_CTYPE, IN_CTB, IN_CBIDX,
  P_EMB0_RM, P_EMB0_RV, P_EMB0_G, P_EMB0_B, P_EMB0_W, P_EMB0_LB,      //  8..13
  P_EMB1_RM, P_EMB1_RV, P_EMB1_G, P_EMB1_B, P_EMB1_W, P_EMB1_LB,      // 14..19
  P_EN0_RM,  P_EN0_RV,  P_EN0_G,  P_EN0_B,  P_EN0_W,  P_EN0_LB,       // 20..25
  P_EN1_RM,  P_EN1_RV,  P_EN1_G,  P_EN1_B,  P_EN1_W,  P_EN1_LB,       // 26..31
  P_EN2_RM,  P_EN2_RV,  P_EN2_G,  P_EN2_B,  P_EN2_W,  P_EN2_LB,       // 32..37
  P_EN3_RM,  P_EN3_RV,  P_EN3_G,  P_EN3_B,  P_EN3_W,  P_EN3_LB,       // 38..43
  P_ROOT, P_NNB, P_GWIH, P_GWHH, P_GBIH, P_GBHH,                      // 44..49
  P_LWIH, P_LWHH, P_LBIH, P_LBHH,                                     // 50..53
  P_FC0_RM, P_FC0_RV, P_FC0_G, P_FC0_B, P_FC0_W, P_FC0_LB,            // 54..59
  P_FC1_RM, P_FC1_RV, P_FC1_G, P_FC1_B, P_FC1_W, P_FC1_LB,            // 60..65
  IN_COUNT
};

extern "C" void kernel_launch(void* const* d_in, const int* in_sizes, int n_in,
                              void* d_out, int out_size, void* d_ws, size_t ws_size,
                              hipStream_t stream) {
  auto F = [&](int i) { return (const float*)d_in[i]; };
  auto I = [&](int i) { return (const int*)d_in[i]; };

  // workspace carve-out
  char* ws = (char*)d_ws;
  size_t cur = 0;
  auto alloc = [&](size_t bytes) -> void* {
    void* p = ws + cur;
    cur = (cur + bytes + 255) & ~(size_t)255;
    return p;
  };
  float* state   = (float*)alloc((size_t)N_ * 32 * 4);
  bf16*  e3n     = (bf16*) alloc((size_t)E_ * 64 * 2);
  bf16*  wbuf    = (bf16*) alloc((size_t)E_ * 1024 * 2);
  float* agg     = (float*)alloc((size_t)N_ * 32 * 4);
  float* deg     = (float*)alloc((size_t)N_ * 4);
  bf16*  w3b     = (bf16*) alloc(1024 * 64 * 2);
  bf16*  fc0b    = (bf16*) alloc(1024 * 160 * 2);
  float* W1p     = (float*)alloc(8 * 1024 * 4);
  float* const1  = (float*)alloc(8 * 4);
  float* q_star  = (float*)alloc((size_t)B_ * 64 * 4);
  float* hs      = (float*)alloc((size_t)B_ * 32 * 4);
  float* cs      = (float*)alloc((size_t)B_ * 32 * 4);
  unsigned* emax = (unsigned*)alloc((size_t)B_ * 4);
  float* denom   = (float*)alloc((size_t)B_ * 4);
  float* ebuf    = (float*)alloc((size_t)N_ * 4);
  float* anum    = (float*)alloc((size_t)N_ * 4);
  bf16*  featsA  = (bf16*) alloc((size_t)C_ * 160 * 2);
  float* out8    = (float*)alloc((size_t)C_ * 8 * 4);
  if (cur > ws_size) return;  // workspace too small; bail deterministically

  const dim3 T(256);

  // weight prep (bf16 conversions + fc1/BN folding)
  k_prep_w3 <<<256, T, 0, stream>>>(F(P_EN3_W), w3b);
  k_prep_fc0<<<640, T, 0, stream>>>(F(P_FC0_W), fc0b);
  k_prep_fc1<<<8,   T, 0, stream>>>(F(P_FC1_W), F(P_FC1_RM), F(P_FC1_RV),
                                    F(P_FC1_G), F(P_FC1_B), F(P_FC1_LB), W1p, const1);

  // node embedding -> state
  k_node_embed<<<N_ * 32 / 256, T, 0, stream>>>(F(IN_NODE),
      F(P_EMB0_RM), F(P_EMB0_RV), F(P_EMB0_G), F(P_EMB0_B), F(P_EMB0_W), F(P_EMB0_LB),
      F(P_EMB1_RM), F(P_EMB1_RV), F(P_EMB1_G), F(P_EMB1_B), F(P_EMB1_W), F(P_EMB1_LB),
      state);

  // edge net front (incl. en3's BN) -> e3n
  EdgeNetParams EP = {
    F(P_EN0_RM), F(P_EN0_RV), F(P_EN0_G), F(P_EN0_B), F(P_EN0_W), F(P_EN0_LB),
    F(P_EN1_RM), F(P_EN1_RV), F(P_EN1_G), F(P_EN1_B), F(P_EN1_W), F(P_EN1_LB),
    F(P_EN2_RM), F(P_EN2_RV), F(P_EN2_G), F(P_EN2_B), F(P_EN2_W), F(P_EN2_LB),
    F(P_EN3_RM), F(P_EN3_RV), F(P_EN3_G), F(P_EN3_B)
  };
  k_edge_net<<<E_ * 32 / 256, T, 0, stream>>>(F(IN_EDGE), EP, e3n);

  // en3 GEMM (WMMA) -> w (bf16)
  k_gemm_en3<<<E_ / 16, T, 0, stream>>>(e3n, w3b, F(P_EN3_LB), wbuf);

  // degrees
  k_zero<<<(N_ + 255) / 256, T, 0, stream>>>(deg, N_);
  k_deg <<<(E_ + 255) / 256, T, 0, stream>>>(I(IN_EIDX), deg);

  // message passing + GRU, 6 steps
  for (int s = 0; s < STEPS_; ++s) {
    k_zero<<<N_ * 32 / 256, T, 0, stream>>>(agg, (long)N_ * 32);
    k_msg <<<E_ * 32 / 256, T, 0, stream>>>(state, wbuf, I(IN_EIDX), agg);
    k_gru <<<N_ * 32 / 256, T, 0, stream>>>(state, agg, deg, F(P_ROOT), F(P_NNB),
                                            F(P_GWIH), F(P_GWHH), F(P_GBIH), F(P_GBHH));
  }

  // Set2Set init
  k_zero<<<B_ * 64 / 256, T, 0, stream>>>(q_star, (long)B_ * 64);
  k_zero<<<B_ * 32 / 256, T, 0, stream>>>(hs, (long)B_ * 32);
  k_zero<<<B_ * 32 / 256, T, 0, stream>>>(cs, (long)B_ * 32);

  // Set2Set, 6 steps
  for (int s = 0; s < STEPS_; ++s) {
    k_lstm<<<B_ * 32 / 256, T, 0, stream>>>(q_star, hs, cs,
                                            F(P_LWIH), F(P_LWHH), F(P_LBIH), F(P_LBHH));
    k_s2s_reset<<<B_ * 32 / 256, T, 0, stream>>>(q_star, emax, denom);
    k_attn_e<<<N_ * 32 / 256, T, 0, stream>>>(state, hs, I(IN_NBIDX), ebuf, emax);
    k_attn_a<<<(N_ + 255) / 256, T, 0, stream>>>(ebuf, emax, I(IN_NBIDX), anum, denom);
    k_attn_r<<<N_ * 32 / 256, T, 0, stream>>>(anum, denom, state, I(IN_NBIDX), q_star);
  }

  // final MLP
  k_feats<<<(int)(((long)C_ * 160) / 256), T, 0, stream>>>(q_star, state,
      I(IN_CBIDX), I(IN_CIDX), I(IN_CTYPE),
      F(P_FC0_RM), F(P_FC0_RV), F(P_FC0_G), F(P_FC0_B), featsA);
  k_zero<<<C_ * 8 / 256, T, 0, stream>>>(out8, (long)C_ * 8);
  k_fc_wmma<<<C_ / 16, T, 0, stream>>>(featsA, fc0b, F(P_FC0_LB), W1p, out8);
  k_final<<<(C_ + 255) / 256, T, 0, stream>>>(out8, I(IN_CTB), const1, (float*)d_out);

  (void)in_sizes; (void)n_in; (void)out_size;
}
